// BasicBlock_39032662786520
// MI455X (gfx1250) — compile-verified, matrix operand order checked by reading, not run on hardware
//
#include <hip/hip_runtime.h>
#include <hip/hip_bf16.h>

// ---------------------------------------------------------------------------
// BasicBlock (per-sample dynamic 3x3 conv x2 + training BN + residual ReLU)
// Implicit-GEMM with v_wmma_f32_16x16x32_bf16, f32 accumulation.
// B=16, C=64, H=W=128.  GEMM per sample: M=64(Cout) x K=576(Cin*9) x N=16384.
// ---------------------------------------------------------------------------

typedef __attribute__((ext_vector_type(8)))  unsigned short ushort8_t;
typedef __attribute__((ext_vector_type(16))) unsigned short ushort16_t;
typedef __attribute__((ext_vector_type(16))) __bf16        v16bf;
typedef __attribute__((ext_vector_type(8)))  float         v8f;

#define B_     16
#define C_     64
#define H_     128
#define W_     128
#define HW_    (H_ * W_)
#define NTOT   (B_ * HW_)          // reduction count per channel = 262144
#define KSLOTS 18                  // 9 taps * 2 ci-blocks of 32
#define CI_STRIDE 72               // padded ci stride in LDS (16B-aligned frags)

__device__ __forceinline__ unsigned short f2bf(float f) {
    unsigned int u = __float_as_uint(f);
    unsigned int r = u + 0x7FFFu + ((u >> 16) & 1u);   // round-to-nearest-even
    return (unsigned short)(r >> 16);
}

// K index held by vector element e for a lane in half {0,1} (ISA 16-bit A/B
// 16x32 layout: elems 0..7 -> K = 8*half + e ; elems 8..15 -> K = 16 + 8*half + (e-8))
__device__ __forceinline__ int k_of_elem(int e, int half) {
    return e + (e & 8) + half * 8;
}

// ---------------------------------------------------------------------------
// Pack filters [B][Cout][Cin][3][3] f32 -> WMMA A fragments, bf16.
// Apack[b][coblk(4)][kslot(18)][lane(32)][elem(16)]
// kslot = (kh*3+kw)*2 + ciblk ; row m = lane%16 ; half = lane/16
// ---------------------------------------------------------------------------
__global__ __launch_bounds__(256) void pack_filters_k(
        const float* __restrict__ F, unsigned short* __restrict__ Apack) {
    int t = blockIdx.x * 256 + threadIdx.x;
    if (t >= B_ * 4 * KSLOTS * 32 * 16) return;
    int e     = t & 15;
    int L     = (t >> 4) & 31;
    int rest  = t >> 9;
    int kslot = rest % KSLOTS;
    int coblk = (rest / KSLOTS) & 3;
    int b     = rest / (KSLOTS * 4);
    int half  = L >> 4, m = L & 15;
    int K     = k_of_elem(e, half);
    int kk    = kslot >> 1;            // kh*3+kw
    int ciblk = kslot & 1;
    int kh = kk / 3, kw = kk % 3;
    int ci = ciblk * 32 + K;
    int co = coblk * 16 + m;
    float v = F[((b * C_ + co) * C_ + ci) * 9 + kh * 3 + kw];
    Apack[t] = f2bf(v);
}

// ---------------------------------------------------------------------------
// One conv layer: per-sample implicit GEMM.
// Grid: (W/64, H, B).  Block: 128 threads = 4 waves; wave w -> Cout block 16w.
// Each wave: 16(co) x 64(n) tile = 4 accumulators of 16x16 f32.
// BN_IN: apply scale/shift + ReLU to input while staging (conv2 path).
// Emits per-channel sum / sumsq via shuffle -> LDS -> global f32 atomics.
// ---------------------------------------------------------------------------
template <bool BN_IN>
__global__ __launch_bounds__(128) void crossconv_wmma_k(
        const float* __restrict__ X,              // [B][C][H][W] f32
        const unsigned short* __restrict__ Apack, // packed bf16 weights
        const float* __restrict__ ab_in,          // scale[64], shift[64]
        float* __restrict__ Out,                  // [B][C][H][W] f32
        float* __restrict__ stats)                // sum[64], sumsq[64] atomics
{
    __shared__ unsigned short Xs[3 * 66 * CI_STRIDE];  // [row 3][col 66][ci 72]
    __shared__ float sred[128];

    const int tid  = threadIdx.x;
    const int lane = tid & 31;
    const int wave = tid >> 5;
    const int half = lane >> 4;
    const int ln   = lane & 15;
    const int w0   = blockIdx.x * 64;
    const int h    = blockIdx.y;
    const int b    = blockIdx.z;

    if (tid < 128) sred[tid] = 0.0f;

    // ---- stage halo patch (rows h-1..h+1, cols w0-1..w0+64, 64 ci) as bf16
    for (int idx = tid; idx < 3 * C_ * 66; idx += 128) {
        int c  = idx % 66;
        int ci = (idx / 66) % C_;
        int r  = idx / (66 * C_);
        int hh = h + r - 1;
        int ww = w0 + c - 1;
        float v = 0.0f;
        if (hh >= 0 && hh < H_ && ww >= 0 && ww < W_) {
            v = X[((b * C_ + ci) * H_ + hh) * W_ + ww];
            if (BN_IN) v = fmaxf(v * ab_in[ci] + ab_in[C_ + ci], 0.0f);
        }
        Xs[(r * 66 + c) * CI_STRIDE + ci] = f2bf(v);
    }
    __syncthreads();

    v8f acc[4];
    {
        v8f z = {0.f, 0.f, 0.f, 0.f, 0.f, 0.f, 0.f, 0.f};
#pragma unroll
        for (int f = 0; f < 4; ++f) acc[f] = z;
    }

    const unsigned short* Abase =
        Apack + (size_t)((b * 4 + wave) * KSLOTS) * 32 * 16;

    // ---- K loop: 9 taps x 2 ci-blocks, K=32 per WMMA
#pragma unroll 1
    for (int kh = 0; kh < 3; ++kh) {
#pragma unroll 1
        for (int kw = 0; kw < 3; ++kw) {
#pragma unroll
            for (int cb = 0; cb < 2; ++cb) {
                int kslot = (kh * 3 + kw) * 2 + cb;
                ushort16_t araw =
                    *(const ushort16_t*)(Abase + (kslot * 32 + lane) * 16);
                v16bf a = __builtin_bit_cast(v16bf, araw);

                int rowbase0 = kh * 66 + kw + ln;  // + f*16 per N fragment
                int cioff    = cb * 32 + half * 8;
#pragma unroll
                for (int f = 0; f < 4; ++f) {
                    const unsigned short* p =
                        &Xs[(rowbase0 + f * 16) * CI_STRIDE + cioff];
                    ushort8_t blo = *(const ushort8_t*)p;          // K 0..7 grp
                    ushort8_t bhi = *(const ushort8_t*)(p + 16);   // K 16..23 grp
                    ushort16_t braw = __builtin_shufflevector(
                        blo, bhi, 0, 1, 2, 3, 4, 5, 6, 7,
                        8, 9, 10, 11, 12, 13, 14, 15);
                    v16bf bb = __builtin_bit_cast(v16bf, braw);
                    acc[f] = __builtin_amdgcn_wmma_f32_16x16x32_bf16(
                        false, a, false, bb, (short)0, acc[f], false, false);
                }
            }
        }
    }

    // ---- epilogue: store f32 tile + per-channel sum / sumsq
    float s[8], q[8];
#pragma unroll
    for (int r = 0; r < 8; ++r) { s[r] = 0.0f; q[r] = 0.0f; }

#pragma unroll
    for (int f = 0; f < 4; ++f) {
#pragma unroll
        for (int r = 0; r < 8; ++r) {
            float v = acc[f][r];
            int co = wave * 16 + r + 8 * half;   // C/D layout: M = r + 8*half
            int w  = w0 + f * 16 + ln;           //             N = lane%16
            Out[((b * C_ + co) * H_ + h) * W_ + w] = v;
            s[r] += v;
            q[r] += v * v;
        }
    }
    // reduce across the 16 lanes sharing each channel (halves stay disjoint)
#pragma unroll
    for (int r = 0; r < 8; ++r) {
#pragma unroll
        for (int off = 1; off < 16; off <<= 1) {
            s[r] += __shfl_xor(s[r], off, 32);
            q[r] += __shfl_xor(q[r], off, 32);
        }
    }
    if (ln == 0) {
#pragma unroll
        for (int r = 0; r < 8; ++r) {
            int co = wave * 16 + r + 8 * half;
            atomicAdd(&sred[co], s[r]);
            atomicAdd(&sred[C_ + co], q[r]);
        }
    }
    __syncthreads();
    if (tid < 128) atomicAdd(&stats[tid], sred[tid]);
}

// ---- fold batch stats into per-channel scale/shift ---------------------------
__global__ void bn_finalize_k(const float* __restrict__ stats,
                              const float* __restrict__ gamma,
                              const float* __restrict__ beta,
                              float* __restrict__ ab) {
    int c = threadIdx.x;
    if (c < C_) {
        const float invN = 1.0f / (float)NTOT;
        float mean = stats[c] * invN;
        float var  = stats[C_ + c] * invN - mean * mean;
        float sc   = gamma[c] * rsqrtf(var + 1e-5f);
        ab[c]      = sc;
        ab[C_ + c] = beta[c] - mean * sc;
    }
}

// ---- out = relu(bn2(conv2) + residual), float4-vectorized -------------------
__global__ __launch_bounds__(256) void residual_relu_k(
        const float4* __restrict__ out2, const float4* __restrict__ x,
        const float* __restrict__ ab2, float4* __restrict__ out, int n4) {
    int i = blockIdx.x * 256 + threadIdx.x;
    if (i >= n4) return;
    int c = (i >> 12) & (C_ - 1);       // 4096 float4 per channel-plane
    float sc = ab2[c], sh = ab2[C_ + c];
    float4 v = out2[i];
    float4 r = x[i];
    float4 o;
    o.x = fmaxf(v.x * sc + sh + r.x, 0.0f);
    o.y = fmaxf(v.y * sc + sh + r.y, 0.0f);
    o.z = fmaxf(v.z * sc + sh + r.z, 0.0f);
    o.w = fmaxf(v.w * sc + sh + r.w, 0.0f);
    out[i] = o;
}

// ---------------------------------------------------------------------------
extern "C" void kernel_launch(void* const* d_in, const int* in_sizes, int n_in,
                              void* d_out, int out_size, void* d_ws, size_t ws_size,
                              hipStream_t stream) {
    const float* x  = (const float*)d_in[0];
    const float* f1 = (const float*)d_in[1];
    const float* f2 = (const float*)d_in[2];
    const float* g1 = (const float*)d_in[3];
    const float* b1 = (const float*)d_in[4];
    const float* g2 = (const float*)d_in[5];
    const float* b2 = (const float*)d_in[6];

    char* ws = (char*)d_ws;
    const size_t featBytes = (size_t)NTOT * C_ / B_ * B_ * sizeof(float); // 67,108,864
    const size_t packBytes = (size_t)B_ * 4 * KSLOTS * 32 * 16 * sizeof(unsigned short); // 1,179,648

    float*          out1 = (float*)ws;
    float*          out2 = (float*)(ws + featBytes);
    unsigned short* Ap1  = (unsigned short*)(ws + 2 * featBytes);
    unsigned short* Ap2  = (unsigned short*)(ws + 2 * featBytes + packBytes);
    float*          st1  = (float*)(ws + 2 * featBytes + 2 * packBytes);
    float*          st2  = st1 + 128;
    float*          ab1  = st1 + 256;
    float*          ab2  = st1 + 384;

    // zero both stats blocks (256 floats) — capture-safe memset node
    hipMemsetAsync(st1, 0, 256 * sizeof(float), stream);

    // pack both filter sets into WMMA A-fragment order
    const int packThreads = B_ * 4 * KSLOTS * 32 * 16;
    pack_filters_k<<<(packThreads + 255) / 256, 256, 0, stream>>>(f1, Ap1);
    pack_filters_k<<<(packThreads + 255) / 256, 256, 0, stream>>>(f2, Ap2);

    dim3 grid(W_ / 64, H_, B_);   // (2, 128, 16) = 4096 workgroups
    crossconv_wmma_k<false><<<grid, 128, 0, stream>>>(x, Ap1, nullptr, out1, st1);
    bn_finalize_k<<<1, 64, 0, stream>>>(st1, g1, b1, ab1);
    crossconv_wmma_k<true><<<grid, 128, 0, stream>>>(out1, Ap2, ab1, out2, st2);
    bn_finalize_k<<<1, 64, 0, stream>>>(st2, g2, b2, ab2);

    const int n4 = NTOT * C_ / 4;   // wait: NTOT already includes B*H*W; elems = B*C*H*W
    // B*C*H*W = 16*64*128*128 = 16,777,216 -> 4,194,304 float4
    const int nf4 = (B_ * C_ * H_ * W_) / 4;
    (void)n4;
    residual_relu_k<<<(nf4 + 255) / 256, 256, 0, stream>>>(
        (const float4*)out2, (const float4*)x, ab2, (float4*)d_out, nf4);
}
